// EdgeDecoder_14474039787539
// MI455X (gfx1250) — matrix operand order repaired and checked by reading, not compiled, not run on hardware
//
#include <hip/hip_runtime.h>

// ---------------------------------------------------------------------------
// EdgeDecoder for MI455X (gfx1250, wave32):
//   out[e] = W2 . relu(W1 @ concat(z_user[row[e]], z_books[col[e]]) + b1) + b2
//
// Phase 1: one-shot fp32 -> bf16 (RNE) conversion of z_user/z_books/W1 into
//          d_ws (~25.6 MB; memory-bound; tables then L2-resident, 192 MB L2).
// Phase 2: 16 edges per wave-iteration as H[64x16] = W1[64x128] * Z[128x16]
//          via v_wmma_f32_16x16x32_bf16 (4 M-tiles x 4 K-chunks = 16 WMMAs).
//          - W1 A-fragments resident in VGPRs for the whole edge loop
//          - B fragments are direct bf16 b128 gathers (no conversion VALU)
//          - b1 bias enters as the C operand of the first WMMA per tile
//          - relu = single v_med3_num_f32 clamp(x, 0, FLT_MAX)
//          - W2 dot lowered to v_pk_fma_f32 by the compiler
//          - 32-bit index/byte offsets everywhere (saddr + scaled voffset)
// ---------------------------------------------------------------------------

typedef __attribute__((ext_vector_type(16))) __bf16        v16bf;
typedef __attribute__((ext_vector_type(8)))  float         v8f;
typedef __attribute__((ext_vector_type(8)))  unsigned int  v8u;

union FragU { v8u u; v16bf b; };

#define HIDDEN 64
#define K2     128   // 2*HIDDEN

// full RNE fp32->bf16 pair pack (one-shot converter only)
__device__ __forceinline__ unsigned int pack2bf_rne(float lo, float hi) {
  unsigned int ul = __float_as_uint(lo);
  unsigned int uh = __float_as_uint(hi);
  ul = (ul + 0x7FFFu + ((ul >> 16) & 1u)) >> 16;
  uh = (((uh + 0x7FFFu + ((uh >> 16) & 1u)) >> 16) << 16);
  return (ul & 0xFFFFu) | uh;
}

// cheap 3-op pack (round-to-nearest ties-away) for the fallback path
__device__ __forceinline__ unsigned int pack2bf_rn(float lo, float hi) {
  unsigned int ul = __float_as_uint(lo) + 0x8000u;
  unsigned int uh = __float_as_uint(hi) + 0x8000u;
  return __builtin_amdgcn_perm(uh, ul, 0x07060302u); // {uh[31:16], ul[31:16]}
}

// relu as a single v_med3_num_f32: clamp(x, 0, FLT_MAX). Finite upper bound
// so InstCombine can't refold into canonicalize+maxnum.
__device__ __forceinline__ float relu1(float x) {
  return __builtin_amdgcn_fmed3f(x, 0.0f, 3.402823466e+38f);
}

// ---------------------------------------------------------------------------
// Phase 1: fp32 -> bf16 converter (grid-stride, 8 floats / thread / iter)
// ---------------------------------------------------------------------------
__global__ void cvt_f32_bf16(const float* __restrict__ src,
                             unsigned int* __restrict__ dst, int nElem) {
  int idx = blockIdx.x * blockDim.x + threadIdx.x;
  int stride = gridDim.x * blockDim.x;
  int nOct = nElem >> 3;                  // sizes are multiples of 8
  for (int i = idx; i < nOct; i += stride) {
    float4 f0 = ((const float4*)src)[2 * i];
    float4 f1 = ((const float4*)src)[2 * i + 1];
    uint4 o;
    o.x = pack2bf_rne(f0.x, f0.y);
    o.y = pack2bf_rne(f0.z, f0.w);
    o.z = pack2bf_rne(f1.x, f1.y);
    o.w = pack2bf_rne(f1.z, f1.w);
    ((uint4*)dst)[i] = o;
  }
}

// ---------------------------------------------------------------------------
// Phase 2 (fast path): bf16 tables in d_ws
// ---------------------------------------------------------------------------
__global__ void __launch_bounds__(256)
edge_decoder_wmma_bf16(const unsigned short* __restrict__ zub, // [U,64] bf16
                       const unsigned short* __restrict__ zbb, // [B,64] bf16
                       const unsigned short* __restrict__ W1b, // [64,128] bf16
                       const unsigned* __restrict__ eidx32,    // [2,E] int64,
                                                               // viewed as u32
                       const float* __restrict__ b1,
                       const float* __restrict__ W2,
                       const float* __restrict__ b2,
                       float* __restrict__ out,
                       int E, int numGroups)
{
  const int lane = threadIdx.x & 31;
  const int n    = lane & 15;
  const int hi   = lane >> 4;
  const int wavesPerBlock = blockDim.x >> 5;
  const int gw   = blockIdx.x * wavesPerBlock + (threadIdx.x >> 5);
  const int totalWaves = gridDim.x * wavesPerBlock;

  // A fragments (W1 bf16), contiguous K order (k = 16*hi + s), same as B, resident in VGPRs.
  // lane m = n; VGPR j 0..7: K = 32c + hi*16 + 2j,2j+1.
  FragU a[4][4];
#pragma unroll
  for (int t = 0; t < 4; ++t) {
    const unsigned short* wrow = W1b + (size_t)(t * 16 + n) * K2;
#pragma unroll
    for (int c = 0; c < 4; ++c) {
      const unsigned short* r1 = wrow + 32 * c + hi * 16;
      uint4 q1 = *(const uint4*)(r1);
      uint4 q2 = *(const uint4*)(r1 + 8);
      a[t][c].u[0] = q1.x; a[t][c].u[1] = q1.y;
      a[t][c].u[2] = q1.z; a[t][c].u[3] = q1.w;
      a[t][c].u[4] = q2.x; a[t][c].u[5] = q2.y;
      a[t][c].u[6] = q2.z; a[t][c].u[7] = q2.w;
    }
  }

  // D VGPR r of tile t holds H[m][n], m = t*16 + hi*8 + r.
  // Bias as persistent C fragments (enters via first WMMA's C operand).
  v8f b1frag[4];
  float w2r[32];
#pragma unroll
  for (int t = 0; t < 4; ++t)
#pragma unroll
    for (int r = 0; r < 8; ++r) {
      int m = t * 16 + hi * 8 + r;
      b1frag[t][r]   = b1[m];
      w2r[t * 8 + r] = W2[m];
    }
  const float b2v = b2[0];

  for (int g = gw; g < numGroups; g += totalWaves) {
    int e = g * 16 + n;
    bool valid = (e < E);
    int ec = valid ? e : (E - 1);
    // int64 indices: little-endian low dword holds the value (< 2^25).
    // 32-bit element index -> saddr + scaled-voffset b32 loads.
    unsigned rowI = eidx32[2u * (unsigned)ec];
    unsigned colI = eidx32[2u * (unsigned)(E + ec)];
    // 32-bit BYTE offsets (row stride = 64 bf16 = 128 B): one v_lshlrev_b32
    unsigned uOff = rowI << 7;
    unsigned bOff = colI << 7;
    const unsigned short* ubase =
        (const unsigned short*)((const char*)zub + uOff);
    const unsigned short* bbase =
        (const unsigned short*)((const char*)zbb + bOff);

    v8f acc[4];

    // ---- chunk 0: C = bias fragments (no acc init movs, no epilogue adds)
    {
      const unsigned short* src = ubase + hi * 16;   // K 0..31 -> user row
      uint4 q0 = ((const uint4*)src)[0];
      uint4 q1 = ((const uint4*)src)[1];
      FragU bf;
      bf.u[0] = q0.x; bf.u[1] = q0.y; bf.u[2] = q0.z; bf.u[3] = q0.w;
      bf.u[4] = q1.x; bf.u[5] = q1.y; bf.u[6] = q1.z; bf.u[7] = q1.w;
#pragma unroll
      for (int t = 0; t < 4; ++t) {
        acc[t] = __builtin_amdgcn_wmma_f32_16x16x32_bf16(
            false, a[t][0].b, false, bf.b, (short)0, b1frag[t], false, false);
      }
    }

    // ---- chunks 1..3: accumulate
#pragma unroll
    for (int c = 1; c < 4; ++c) {
      int kb = 32 * c + hi * 16;
      const unsigned short* src =
          (kb < HIDDEN) ? (ubase + kb) : (bbase + (kb - HIDDEN));
      uint4 q0 = ((const uint4*)src)[0];
      uint4 q1 = ((const uint4*)src)[1];
      FragU bf;
      bf.u[0] = q0.x; bf.u[1] = q0.y; bf.u[2] = q0.z; bf.u[3] = q0.w;
      bf.u[4] = q1.x; bf.u[5] = q1.y; bf.u[6] = q1.z; bf.u[7] = q1.w;
#pragma unroll
      for (int t = 0; t < 4; ++t) {
        acc[t] = __builtin_amdgcn_wmma_f32_16x16x32_bf16(
            false, a[t][c].b, false, bf.b, (short)0, acc[t], false, false);
      }
    }

    // ---- epilogue: relu (v_med3) + W2 dot (v_pk_fma_f32), 4 chains
    float p0 = 0.f, p1 = 0.f, p2 = 0.f, p3 = 0.f;
#pragma unroll
    for (int r = 0; r < 8; ++r) {
      p0 = fmaf(relu1(acc[0][r]), w2r[r],      p0);
      p1 = fmaf(relu1(acc[1][r]), w2r[8 + r],  p1);
      p2 = fmaf(relu1(acc[2][r]), w2r[16 + r], p2);
      p3 = fmaf(relu1(acc[3][r]), w2r[24 + r], p3);
    }
    float partial = (p0 + p1) + (p2 + p3);
    partial += __shfl_xor(partial, 16, 32);
    if (hi == 0 && valid) out[e] = partial + b2v;
  }
}

// ---------------------------------------------------------------------------
// Fallback (ws too small): fp32 gather + cheap in-register pack
// ---------------------------------------------------------------------------
__global__ void __launch_bounds__(256)
edge_decoder_wmma_f32(const float* __restrict__ zu, const float* __restrict__ zb,
                      const unsigned* __restrict__ eidx32,
                      const float* __restrict__ W1, const float* __restrict__ b1,
                      const float* __restrict__ W2, const float* __restrict__ b2,
                      float* __restrict__ out, int E, int numGroups)
{
  const int lane = threadIdx.x & 31;
  const int n    = lane & 15;
  const int hi   = lane >> 4;
  const int wavesPerBlock = blockDim.x >> 5;
  const int gw   = blockIdx.x * wavesPerBlock + (threadIdx.x >> 5);
  const int totalWaves = gridDim.x * wavesPerBlock;

  FragU a[4][4];
#pragma unroll
  for (int t = 0; t < 4; ++t) {
    const float* wrow = W1 + (size_t)(t * 16 + n) * K2;
#pragma unroll
    for (int c = 0; c < 4; ++c) {
      const float* r1 = wrow + 32 * c + hi * 16;
      const float* r2 = r1 + 8;
      float4 x0 = *(const float4*)(r1);
      float4 x1 = *(const float4*)(r1 + 4);
      float4 x2 = *(const float4*)(r2);
      float4 x3 = *(const float4*)(r2 + 4);
      a[t][c].u[0] = pack2bf_rn(x0.x, x0.y);
      a[t][c].u[1] = pack2bf_rn(x0.z, x0.w);
      a[t][c].u[2] = pack2bf_rn(x1.x, x1.y);
      a[t][c].u[3] = pack2bf_rn(x1.z, x1.w);
      a[t][c].u[4] = pack2bf_rn(x2.x, x2.y);
      a[t][c].u[5] = pack2bf_rn(x2.z, x2.w);
      a[t][c].u[6] = pack2bf_rn(x3.x, x3.y);
      a[t][c].u[7] = pack2bf_rn(x3.z, x3.w);
    }
  }

  v8f b1frag[4];
  float w2r[32];
#pragma unroll
  for (int t = 0; t < 4; ++t)
#pragma unroll
    for (int r = 0; r < 8; ++r) {
      int m = t * 16 + hi * 8 + r;
      b1frag[t][r]   = b1[m];
      w2r[t * 8 + r] = W2[m];
    }
  const float b2v = b2[0];

  for (int g = gw; g < numGroups; g += totalWaves) {
    int e = g * 16 + n;
    bool valid = (e < E);
    int ec = valid ? e : (E - 1);
    unsigned rowI = eidx32[2u * (unsigned)ec];
    unsigned colI = eidx32[2u * (unsigned)(E + ec)];
    unsigned uOff = rowI << 8;   // 64 floats = 256 B rows
    unsigned bOff = colI << 8;
    const float* ubase = (const float*)((const char*)zu + uOff);
    const float* bbase = (const float*)((const char*)zb + bOff);

    v8f acc[4];
#pragma unroll
    for (int c = 0; c < 4; ++c) {
      int kb = 32 * c + hi * 16;
      const float* src = (kb < HIDDEN) ? (ubase + kb) : (bbase + (kb - HIDDEN));
      float4 y0 = *(const float4*)(src);
      float4 y1 = *(const float4*)(src + 4);
      float4 y2 = *(const float4*)(src + 8);
      float4 y3 = *(const float4*)(src + 12);
      FragU bf;
      bf.u[0] = pack2bf_rn(y0.x, y0.y);
      bf.u[1] = pack2bf_rn(y0.z, y0.w);
      bf.u[2] = pack2bf_rn(y1.x, y1.y);
      bf.u[3] = pack2bf_rn(y1.z, y1.w);
      bf.u[4] = pack2bf_rn(y2.x, y2.y);
      bf.u[5] = pack2bf_rn(y2.z, y2.w);
      bf.u[6] = pack2bf_rn(y3.x, y3.y);
      bf.u[7] = pack2bf_rn(y3.z, y3.w);
#pragma unroll
      for (int t = 0; t < 4; ++t) {
        acc[t] = __builtin_amdgcn_wmma_f32_16x16x32_bf16(
            false, a[t][c].b, false, bf.b, (short)0,
            (c == 0) ? b1frag[t] : acc[t], false, false);
      }
    }

    float p0 = 0.f, p1 = 0.f, p2 = 0.f, p3 = 0.f;
#pragma unroll
    for (int r = 0; r < 8; ++r) {
      p0 = fmaf(relu1(acc[0][r]), w2r[r],      p0);
      p1 = fmaf(relu1(acc[1][r]), w2r[8 + r],  p1);
      p2 = fmaf(relu1(acc[2][r]), w2r[16 + r], p2);
      p3 = fmaf(relu1(acc[3][r]), w2r[24 + r], p3);
    }
    float partial = (p0 + p1) + (p2 + p3);
    partial += __shfl_xor(partial, 16, 32);
    if (hi == 0 && valid) out[e] = partial + b2v;
  }
}

extern "C" void kernel_launch(void* const* d_in, const int* in_sizes, int n_in,
                              void* d_out, int out_size, void* d_ws, size_t ws_size,
                              hipStream_t stream) {
  (void)n_in;
  const float*    zu     = (const float*)d_in[0];
  const float*    zb     = (const float*)d_in[1];
  const unsigned* eidx32 = (const unsigned*)d_in[2];  // int64 viewed as u32 pairs
  const float*    W1     = (const float*)d_in[3];
  const float*    b1     = (const float*)d_in[4];
  const float*    W2     = (const float*)d_in[5];
  const float*    b2     = (const float*)d_in[6];
  float*          out    = (float*)d_out;

  const int E = out_size;
  const int numGroups = (E + 15) / 16;

  const int threads = 256;
  const int wavesPerBlock = threads / 32;
  int blocks = 1024;                      // ~8192 waves, ~15 groups/wave
  int maxBlocks = (numGroups + wavesPerBlock - 1) / wavesPerBlock;
  if (blocks > maxBlocks) blocks = maxBlocks;

  const size_t nU = (size_t)in_sizes[0];  // NUM_USERS * 64
  const size_t nB = (size_t)in_sizes[1];  // NUM_BOOKS * 64
  const size_t nW = (size_t)in_sizes[3];  // 64 * 128
  const size_t need = (nU + nB + nW) * sizeof(unsigned short);

  if (ws_size >= need) {
    unsigned short* zub = (unsigned short*)d_ws;
    unsigned short* zbb = zub + nU;
    unsigned short* W1b = zbb + nB;

    int cb;
    cb = (int)((nU / 8 + 255) / 256);
    cvt_f32_bf16<<<cb, 256, 0, stream>>>(zu, (unsigned int*)zub, (int)nU);
    cb = (int)((nB / 8 + 255) / 256);
    cvt_f32_bf16<<<cb, 256, 0, stream>>>(zb, (unsigned int*)zbb, (int)nB);
    cb = (int)((nW / 8 + 255) / 256);
    cvt_f32_bf16<<<cb, 256, 0, stream>>>(W1, (unsigned int*)W1b, (int)nW);

    edge_decoder_wmma_bf16<<<blocks, threads, 0, stream>>>(
        zub, zbb, W1b, eidx32, b1, W2, b2, out, E, numGroups);
  } else {
    edge_decoder_wmma_f32<<<blocks, threads, 0, stream>>>(
        zu, zb, eidx32, W1, b1, W2, b2, out, E, numGroups);
  }
}
